// KAGIN_cls_36051955483208
// MI455X (gfx1250) — compile-verified, matrix operand order checked by reading, not run on hardware
//
#include <hip/hip_runtime.h>
#include <hip/hip_fp16.h>
#include <math.h>
#include <stdint.h>

typedef __attribute__((ext_vector_type(16))) _Float16 v16h;
typedef __attribute__((ext_vector_type(8)))  _Float16 v8h;
typedef __attribute__((ext_vector_type(8)))  float    v8f;

#define NN      50000
#define NE      800000
#define F_IN    64
#define HID     128
#define NCLS    10
#define NCLSP   16
#define BN_EPS_ 1e-5f

// ---------------------------------------------------------------- utilities
__global__ void fill_zero_f32(float* p, int n) {
    int i = blockIdx.x * blockDim.x + threadIdx.x;
    if (i < n) p[i] = 0.f;
}

__global__ void copy_f32(const float* __restrict__ s, float* __restrict__ d, int n) {
    int i = blockIdx.x * blockDim.x + threadIdx.x;
    if (i < n) d[i] = s[i];
}

// ------------------------------------------------- GIN neighbor aggregation
// agg[dst] += x[src]; agg pre-initialized with x (eps=0 GIN).
__global__ void scatter_add_f32(const float* __restrict__ x, const int* __restrict__ ei,
                                float* __restrict__ agg, int nfeat) {
    int groups = nfeat >> 2;                       // float4 groups per edge
    long long tid = (long long)blockIdx.x * blockDim.x + threadIdx.x;
    long long total = (long long)NE * groups;
    if (tid >= total) return;
    int e = (int)(tid / groups);
    int c = (int)(tid % groups) << 2;
    int s = ei[e];
    int d = ei[NE + e];
    float4 v = *(const float4*)(x + (size_t)s * nfeat + c);
    float* o = agg + (size_t)d * nfeat + c;
    atomicAdd(o + 0, v.x);
    atomicAdd(o + 1, v.y);
    atomicAdd(o + 2, v.z);
    atomicAdd(o + 3, v.w);
}

// --------------------------------------- KAN activation: silu + B-splines
// Writes f16 A[N, K] row-major, K = infeat*9:
//   cols [0, infeat)           : silu(x_i)
//   cols [infeat + 8i, +8)     : Cox-de Boor bases (order 3, uniform knots)
__global__ void kan_activate(const float* __restrict__ xin, _Float16* __restrict__ A, int infeat) {
    int tid = blockIdx.x * blockDim.x + threadIdx.x;
    int total = NN * infeat;
    if (tid >= total) return;
    int n = tid / infeat, i = tid % infeat;
    int K = infeat * 9;
    float x = xin[(size_t)n * infeat + i];

    float b[11];
#pragma unroll
    for (int j = 0; j < 11; ++j) {
        float t0 = -2.2f + 0.4f * j;
        b[j] = (x >= t0 && x < t0 + 0.4f) ? 1.f : 0.f;
    }
#pragma unroll
    for (int k = 1; k <= 3; ++k) {
        float inv = 1.f / (0.4f * k);
#pragma unroll
        for (int j = 0; j < 11; ++j) {
            if (j < 11 - k) {
                float tj = -2.2f + 0.4f * j;
                float l = (x - tj) * inv * b[j];
                float r = ((tj + 0.4f * (k + 1)) - x) * inv * b[j + 1];
                b[j] = l + r;
            }
        }
    }
    float s = x / (1.f + __expf(-x));   // silu

    _Float16* row = A + (size_t)n * K;
    row[i] = (_Float16)s;
    v8h bv;
#pragma unroll
    for (int g = 0; g < 8; ++g) bv[g] = (_Float16)b[g];
    *(v8h*)(row + infeat + i * 8) = bv;   // 16B aligned (infeat even, i*8)
}

// ------------------------- fold base_w + spline_w*scaler into swizzled f16 B
// Bswz laid out per (kt, nt) WMMA fragment: 32 lanes x 16 contiguous halves,
// lane = col%16 + 16*hi, element e -> kk = (e<8 ? hi*8+e : 16+hi*8+e-8).
__global__ void prep_weights(const float* __restrict__ bw, const float* __restrict__ sw,
                             const float* __restrict__ sc, _Float16* __restrict__ Bswz,
                             int infeat, int outfeat, int outp) {
    int K = infeat * 9;
    int NT = outp >> 4;
    int total = K * outp;
    int tid = blockIdx.x * blockDim.x + threadIdx.x;
    if (tid >= total) return;
    int e = tid & 15;
    int lanei = (tid >> 4) & 31;
    int blk = tid >> 9;              // kt*NT + nt
    int nt = blk % NT, kt = blk / NT;
    int nn = lanei & 15, hi = lanei >> 4;
    int kk = (e < 8) ? (hi * 8 + e) : (16 + hi * 8 + (e - 8));
    int k = kt * 32 + kk;
    int o = nt * 16 + nn;
    float val = 0.f;
    if (o < outfeat && k < K) {
        if (k < infeat) {
            val = bw[(size_t)o * infeat + k];
        } else {
            int i = (k - infeat) >> 3, g = (k - infeat) & 7;
            val = sw[((size_t)o * infeat + i) * 8 + g] * sc[(size_t)o * infeat + i];
        }
    }
    Bswz[tid] = (_Float16)val;
}

// ----------------------------------------------------- WMMA GEMM  C = A x B
// 256 threads = 8 waves; each wave owns 16 rows x (NT*16) cols; block covers
// 128 rows. B is staged into LDS with GLOBAL_LOAD_ASYNC_TO_LDS_B128 (ASYNCcnt,
// no VGPR round-trip), double-buffered in 2-ktile (K=64) chunks so the async
// DMA of the next chunk overlaps WMMA on the current chunk.
template <int NT>
__global__ __launch_bounds__(256) void kan_gemm_wmma(const _Float16* __restrict__ A,
                                                     const _Float16* __restrict__ Bswz,
                                                     float* __restrict__ C,
                                                     int K, int ncols, int ldc) {
    constexpr int CHUNK_HALVES = NT * 1024;          // 2 ktiles * NT * 32 lanes * 16 halves
    constexpr int CHUNK_BYTES  = CHUNK_HALVES * 2;
    constexpr int UNITS = NT * 128;                  // 16-byte units per chunk
    constexpr int MAXCP = (UNITS + 255) / 256;       // async issues per thread
    __shared__ _Float16 lds[2 * CHUNK_HALVES];

    int tid = threadIdx.x;
    int wave = tid >> 5;
    int lane = tid & 31;
    int r0 = blockIdx.x * 128 + wave * 16;
    int row = r0 + (lane & 15);
    int rowc = row < NN ? row : NN - 1;      // clamp: EXEC must stay all-ones for WMMA
    int hi = lane >> 4;

    v8f acc[NT];
#pragma unroll
    for (int t = 0; t < NT; ++t)
#pragma unroll
        for (int j = 0; j < 8; ++j) acc[t][j] = 0.f;

    const _Float16* arow = A + (size_t)rowc * K;
    int nchunks = K >> 6;                    // K is a multiple of 64 (576, 1152)

    // flat shared pointer: low 32 bits are the LDS byte offset (aperture rule)
    unsigned lds_base = (unsigned)(uintptr_t)(void*)&lds[0];

    auto issue_chunk = [&](int c, int buf) {
        const char* src = (const char*)Bswz + (size_t)c * CHUNK_BYTES;
        unsigned lb = lds_base + (unsigned)buf * CHUNK_BYTES;
#pragma unroll
        for (int i = 0; i < MAXCP; ++i) {
            int u = tid + i * 256;
            if (u < UNITS) {
                unsigned long long ga = (unsigned long long)(uintptr_t)(src + u * 16);
                unsigned la = lb + (unsigned)u * 16;
                asm volatile("global_load_async_to_lds_b128 %0, %1, off"
                             :: "v"(la), "v"(ga) : "memory");
            }
        }
    };

    // prologue: DMA chunk 0 into buffer 0
    issue_chunk(0, 0);
    asm volatile("s_wait_asynccnt 0x0" ::: "memory");
    __syncthreads();

    for (int c = 0; c < nchunks; ++c) {
        // DMA next chunk into the other buffer (free: barrier at end of c-1
        // guaranteed all waves finished reading it)
        if (c + 1 < nchunks) {
            issue_chunk(c + 1, (c + 1) & 1);
            // warm cache for next A chunk too (global_prefetch_b8)
            __builtin_prefetch(arow + (size_t)(c + 1) * 64, 0, 0);
        }

        const _Float16* bb = lds + (c & 1) * CHUNK_HALVES;
#pragma unroll
        for (int kl = 0; kl < 2; ++kl) {
            int kt = c * 2 + kl;
            v8h lo = *(const v8h*)(arow + kt * 32 + hi * 8);
            v8h hh = *(const v8h*)(arow + kt * 32 + 16 + hi * 8);
            v16h a = __builtin_shufflevector(lo, hh, 0, 1, 2, 3, 4, 5, 6, 7,
                                             8, 9, 10, 11, 12, 13, 14, 15);
#pragma unroll
            for (int nt = 0; nt < NT; ++nt) {
                v16h b = *(const v16h*)(bb + ((kl * NT + nt) * 32 + lane) * 16);
                acc[nt] = __builtin_amdgcn_wmma_f32_16x16x32_f16(
                    false, a, false, b, (short)0, acc[nt], false, false);
            }
        }
        if (c + 1 < nchunks)
            asm volatile("s_wait_asynccnt 0x0" ::: "memory");   // own DMA landed
        __syncthreads();                                        // publish to all waves
    }

#pragma unroll
    for (int nt = 0; nt < NT; ++nt) {
        int col = nt * 16 + (lane & 15);
        if (col < ncols) {
#pragma unroll
            for (int r = 0; r < 8; ++r) {
                int rr = r0 + 8 * hi + r;    // C layout: VGPR r -> M = r + 8*hi
                if (rr < NN) C[(size_t)rr * ldc + col] = acc[nt][r];
            }
        }
    }
}

// ---------------------------------------------------------------- BatchNorm
__global__ void bn_stats(const float* __restrict__ h, float* __restrict__ stats) {
    int f = threadIdx.x;                     // 128 threads = 128 features
    int n0 = blockIdx.x * 256;
    int n1 = n0 + 256;
    if (n1 > NN) n1 = NN;
    float s = 0.f, s2 = 0.f;
    for (int n = n0; n < n1; ++n) {
        float v = h[(size_t)n * HID + f];
        s += v;
        s2 += v * v;
    }
    atomicAdd(stats + f, s);
    atomicAdd(stats + HID + f, s2);
}

__global__ void bn_apply(float* __restrict__ h, const float* __restrict__ stats,
                         const float* __restrict__ g, const float* __restrict__ bta) {
    int tid = blockIdx.x * blockDim.x + threadIdx.x;
    if (tid >= NN * HID) return;
    int f = tid & (HID - 1);
    float mean = stats[f] * (1.f / NN);
    float var = stats[HID + f] * (1.f / NN) - mean * mean;
    float v = h[tid];
    h[tid] = (v - mean) * rsqrtf(var + BN_EPS_) * g[f] + bta[f];
}

// -------------------------------------------------------------- log-softmax
__global__ void log_softmax10(const float* __restrict__ z, float* __restrict__ out) {
    int n = blockIdx.x * blockDim.x + threadIdx.x;
    if (n >= NN) return;
    float v[NCLS];
    float m = -1e30f;
#pragma unroll
    for (int c = 0; c < NCLS; ++c) {
        v[c] = z[(size_t)n * NCLSP + c];
        m = fmaxf(m, v[c]);
    }
    float s = 0.f;
#pragma unroll
    for (int c = 0; c < NCLS; ++c) s += __expf(v[c] - m);
    float ls = m + __logf(s);
#pragma unroll
    for (int c = 0; c < NCLS; ++c) out[(size_t)n * NCLS + c] = v[c] - ls;
}

// ------------------------------------------------------------------- driver
extern "C" void kernel_launch(void* const* d_in, const int* in_sizes, int n_in,
                              void* d_out, int out_size, void* d_ws, size_t ws_size,
                              hipStream_t stream) {
    (void)in_sizes; (void)n_in; (void)out_size; (void)ws_size;

    const float* x0 = (const float*)d_in[0];
    const int*   ei = (const int*)d_in[1];
    // d_in[2] = batch (unused on node-level path)
    // kans flattened at 3..26: layer L (0..3), sub s (0..1): (bw, sw, sc)
    auto BW = [&](int L, int s) { return (const float*)d_in[3 + (L * 2 + s) * 3 + 0]; };
    auto SWp = [&](int L, int s) { return (const float*)d_in[3 + (L * 2 + s) * 3 + 1]; };
    auto SCp = [&](int L, int s) { return (const float*)d_in[3 + (L * 2 + s) * 3 + 2]; };
    const float *bng[3], *bnb[3];
    for (int i = 0; i < 3; ++i) {
        bng[i] = (const float*)d_in[27 + 2 * i];
        bnb[i] = (const float*)d_in[27 + 2 * i + 1];
    }

    // workspace partition
    char* ws = (char*)d_ws;
    size_t off = 0;
    auto wsalloc = [&](size_t bytes) {
        void* p = ws + off;
        off = (off + bytes + 255) & ~(size_t)255;
        return p;
    };
    float*     agg   = (float*)wsalloc((size_t)NN * HID * 4);
    float*     h1    = (float*)wsalloc((size_t)NN * HID * 4);
    float*     h2    = (float*)wsalloc((size_t)NN * HID * 4);
    _Float16*  Abuf  = (_Float16*)wsalloc((size_t)NN * (HID * 9) * 2);
    _Float16*  Wbuf  = (_Float16*)wsalloc((size_t)(HID * 9) * HID * 2);
    float*     cls   = (float*)wsalloc((size_t)NN * NCLSP * 4);
    float*     stats = (float*)wsalloc(256 * 4);

    auto run_kan_sub = [&](const float* in, int infeat, int L, int s,
                           int outfeat, int outp, float* outbuf) {
        int K = infeat * 9;
        int nact = NN * infeat;
        kan_activate<<<(nact + 255) / 256, 256, 0, stream>>>(in, Abuf, infeat);
        int wtot = K * outp;
        prep_weights<<<(wtot + 255) / 256, 256, 0, stream>>>(
            BW(L, s), SWp(L, s), SCp(L, s), Wbuf, infeat, outfeat, outp);
        int blocks = (NN + 127) / 128;
        if (outp == 128)
            kan_gemm_wmma<8><<<blocks, 256, 0, stream>>>(Abuf, Wbuf, outbuf, K, outfeat, outp);
        else
            kan_gemm_wmma<1><<<blocks, 256, 0, stream>>>(Abuf, Wbuf, outbuf, K, outfeat, outp);
    };

    const float* xcur = x0;
    int feat = F_IN;
    for (int L = 0; L < 3; ++L) {
        int nel = NN * feat;
        copy_f32<<<(nel + 255) / 256, 256, 0, stream>>>(xcur, agg, nel);
        long long sthreads = (long long)NE * (feat / 4);
        scatter_add_f32<<<(int)((sthreads + 255) / 256), 256, 0, stream>>>(xcur, ei, agg, feat);
        run_kan_sub(agg, feat, L, 0, HID, 128, h1);
        run_kan_sub(h1, HID, L, 1, HID, 128, h2);
        fill_zero_f32<<<1, 256, 0, stream>>>(stats, 256);
        bn_stats<<<(NN + 255) / 256, 128, 0, stream>>>(h2, stats);
        bn_apply<<<(NN * HID + 255) / 256, 256, 0, stream>>>(h2, stats, bng[L], bnb[L]);
        xcur = h2;
        feat = HID;
    }
    // classifier KAN [128 -> 128 -> 10 (padded 16)]
    run_kan_sub(h2, HID, 3, 0, HID, 128, h1);
    run_kan_sub(h1, HID, 3, 1, NCLS, NCLSP, cls);
    log_softmax10<<<(NN + 255) / 256, 256, 0, stream>>>(cls, (float*)d_out);
}
